// ST_GAT_24489903521816
// MI455X (gfx1250) — compile-verified, matrix operand order checked by reading, not run on hardware
//
#include <hip/hip_runtime.h>
#include <math.h>

// ---------------- problem constants ----------------
#define NB    256           // batch graphs
#define TT    90            // timesteps
#define KJ    17            // joints
#define NPER  1530          // nodes per graph
#define NNODE (NB*NPER)     // 391680
#define HH    64            // GAT hidden
#define LHID  512           // LSTM hidden
#define GATE  (4*LHID)      // 2048
#define MTOT  (NB*TT)       // 23040 sequence rows
#define IN1   1122
#define IN1P  1152          // padded K for WMMA (mult of 64)
#define OUTD  57
#define KC    64            // K-chunk staged in LDS per GEMM step

typedef _Float16 h16;
typedef __attribute__((ext_vector_type(16))) _Float16 v16h;
typedef __attribute__((ext_vector_type(8)))  _Float16 v8h;
typedef __attribute__((ext_vector_type(8)))  float    v8f;

// ---------------- device helpers ----------------
__device__ __forceinline__ unsigned fenc(float f) {
  unsigned u = __float_as_uint(f);
  return (u & 0x80000000u) ? ~u : (u | 0x80000000u);
}
__device__ __forceinline__ float fdec(unsigned k) {
  unsigned u = (k & 0x80000000u) ? (k & 0x7fffffffu) : ~k;
  return __uint_as_float(u);
}
__device__ __forceinline__ float sigf(float x) { return 1.0f / (1.0f + __expf(-x)); }

// Async 16B global->LDS copy (CDNA5, tracked by ASYNCcnt).
// LDS byte address == low 32 bits of the generic pointer (ISA aperture truncation).
__device__ __forceinline__ void async_copy_b128(const h16* g, h16* l) {
  unsigned lo = (unsigned)(unsigned long long)l;
  asm volatile("global_load_async_to_lds_b128 %0, %1, off"
               :: "v"(lo), "v"(g) : "memory");
}
__device__ __forceinline__ void wait_asynccnt0() {
#if __has_builtin(__builtin_amdgcn_s_wait_asynccnt)
  __builtin_amdgcn_s_wait_asynccnt(0);
#else
  asm volatile("s_wait_asynccnt 0x0" ::: "memory");
#endif
}

// Build a v16h A/B operand from two 16B chunks:
// elements 0..7  = K[base     + half*8 .. +7]   (contiguous)
// elements 8..15 = K[base+16  + half*8 .. +7]   (contiguous)
__device__ __forceinline__ v16h make_op(const h16* lo_p, const h16* hi_p) {
  v8h lo = *(const v8h*)lo_p;
  v8h hi = *(const v8h*)hi_p;
  v16h r;
#pragma unroll
  for (int i = 0; i < 8; ++i) { r[i] = lo[i]; r[8 + i] = hi[i]; }
  return r;
}
__device__ __forceinline__ v8f wmma32(v16h a, v16h b, v8f c) {
  return __builtin_amdgcn_wmma_f32_16x16x32_f16(false, a, false, b, (short)0, c, false, false);
}

// ---------------- GAT kernels (fp32, memory/atomic bound) ----------------
__global__ void k_xp_first(const float* __restrict__ x, const float* __restrict__ W,
                           float* __restrict__ xp) {
  long long i = (long long)blockIdx.x * blockDim.x + threadIdx.x;
  if (i >= (long long)NNODE * HH) return;
  long long n = i >> 6; int h = (int)(i & 63);
  xp[i] = x[n * 2] * W[h] + x[n * 2 + 1] * W[HH + h];
}

__global__ void k_xp_dense64(const float* __restrict__ hin, const float* __restrict__ W,
                             float* __restrict__ xp) {
  long long i = (long long)blockIdx.x * blockDim.x + threadIdx.x;
  if (i >= (long long)NNODE * HH) return;
  long long n = i >> 6; int h = (int)(i & 63);
  const float* row = hin + n * HH;
  float s = 0.f;
#pragma unroll 8
  for (int k = 0; k < HH; ++k) s += row[k] * W[k * HH + h];
  xp[i] = s;
}

__global__ void k_node_dots(const float* __restrict__ xp, const float* __restrict__ asrc,
                            const float* __restrict__ adst,
                            float* __restrict__ esrc, float* __restrict__ edst) {
  int n = blockIdx.x * blockDim.x + threadIdx.x;
  if (n >= NNODE) return;
  const float* row = xp + (size_t)n * HH;
  float s = 0.f, d = 0.f;
#pragma unroll 8
  for (int i = 0; i < HH; ++i) { float v = row[i]; s += v * asrc[i]; d += v * adst[i]; }
  esrc[n] = s; edst[n] = d;
}

__global__ void k_edge_logit_max(const int* __restrict__ src, const int* __restrict__ dst,
                                 const float* __restrict__ esrc, const float* __restrict__ edst,
                                 float* __restrict__ ebuf, unsigned* __restrict__ maxb, int E) {
  int e = blockIdx.x * blockDim.x + threadIdx.x;
  if (e >= E) return;
  int s = src[e], d = dst[e];
  float z = esrc[s] + edst[d];
  z = (z > 0.f) ? z : 0.2f * z;              // leaky_relu(0.2)
  ebuf[e] = z;
  atomicMax(&maxb[d], fenc(z));
}
__global__ void k_self_logit_max(const float* __restrict__ esrc, const float* __restrict__ edst,
                                 float* __restrict__ eself, unsigned* __restrict__ maxb) {
  int n = blockIdx.x * blockDim.x + threadIdx.x;
  if (n >= NNODE) return;
  float z = esrc[n] + edst[n];
  z = (z > 0.f) ? z : 0.2f * z;
  eself[n] = z;
  atomicMax(&maxb[n], fenc(z));
}
__global__ void k_edge_den(const int* __restrict__ dst, const float* __restrict__ ebuf,
                           const unsigned* __restrict__ maxb, float* __restrict__ den, int E) {
  int e = blockIdx.x * blockDim.x + threadIdx.x;
  if (e >= E) return;
  int d = dst[e];
  atomicAdd(&den[d], __expf(ebuf[e] - fdec(maxb[d])));
}
__global__ void k_self_den(const float* __restrict__ eself, const unsigned* __restrict__ maxb,
                           float* __restrict__ den) {
  int n = blockIdx.x * blockDim.x + threadIdx.x;
  if (n >= NNODE) return;
  atomicAdd(&den[n], __expf(eself[n] - fdec(maxb[n])));
}
__global__ void k_edge_scatter(const int* __restrict__ src, const int* __restrict__ dst,
                               const float* __restrict__ ebuf, const unsigned* __restrict__ maxb,
                               const float* __restrict__ den, const float* __restrict__ xp,
                               float* __restrict__ out, long long EC) {
  long long i = (long long)blockIdx.x * blockDim.x + threadIdx.x;
  if (i >= EC) return;
  int e = (int)(i >> 6), ch = (int)(i & 63);
  int s = src[e], d = dst[e];
  float alpha = __expf(ebuf[e] - fdec(maxb[d])) / den[d];
  atomicAdd(&out[(size_t)d * HH + ch], xp[(size_t)s * HH + ch] * alpha);
}
__global__ void k_self_scatter(const float* __restrict__ eself, const unsigned* __restrict__ maxb,
                               const float* __restrict__ den, const float* __restrict__ xp,
                               float* __restrict__ out) {
  long long i = (long long)blockIdx.x * blockDim.x + threadIdx.x;
  if (i >= (long long)NNODE * HH) return;
  int n = (int)(i >> 6);
  float alpha = __expf(eself[n] - fdec(maxb[n])) / den[n];
  atomicAdd(&out[i], xp[i] * alpha);
}

__global__ void k_bn_stats(const float* __restrict__ v, float* __restrict__ stats) {
  const int ch = threadIdx.x;   // 64
  const int ty = threadIdx.y;   // 4
  float s = 0.f, q = 0.f;
  for (int n = blockIdx.x * 4 + ty; n < NNODE; n += gridDim.x * 4) {
    float x = v[(size_t)n * HH + ch];
    s += x; q += x * x;
  }
  __shared__ float sh[4][HH], shq[4][HH];
  sh[ty][ch] = s; shq[ty][ch] = q;
  __syncthreads();
  if (ty == 0) {
    s = sh[0][ch] + sh[1][ch] + sh[2][ch] + sh[3][ch];
    q = shq[0][ch] + shq[1][ch] + shq[2][ch] + shq[3][ch];
    atomicAdd(&stats[ch], s);
    atomicAdd(&stats[HH + ch], q);
  }
}
__global__ void k_bn_apply_relu(const float* __restrict__ v, const float* __restrict__ stats,
                                const float* __restrict__ g, const float* __restrict__ b,
                                float* __restrict__ out) {
  long long i = (long long)blockIdx.x * blockDim.x + threadIdx.x;
  if (i >= (long long)NNODE * HH) return;
  int ch = (int)(i & 63);
  float mu  = stats[ch] * (1.0f / NNODE);
  float var = stats[HH + ch] * (1.0f / NNODE) - mu * mu;
  float y = g[ch] * (v[i] - mu) * rsqrtf(var + 1e-5f) + b[ch];
  out[i] = fmaxf(y, 0.f);
}

// ---------------- format conversion kernels ----------------
__global__ void k_cvt_pad(const float* __restrict__ s, h16* __restrict__ d,
                          int K, int Kp, long long total) {
  long long i = (long long)blockIdx.x * blockDim.x + threadIdx.x;
  if (i >= total) return;
  long long r = i / Kp; int c = (int)(i % Kp);
  d[i] = (c < K) ? (h16)s[r * K + c] : (h16)0.f;
}
__global__ void k_cvt_T(const float* __restrict__ s /*[K][Nn]*/, h16* __restrict__ d /*[Nn][K]*/,
                        int K, int Nn) {
  long long i = (long long)blockIdx.x * blockDim.x + threadIdx.x;
  if (i >= (long long)Nn * K) return;
  int n = (int)(i / K), k = (int)(i % K);
  d[i] = (h16)s[(size_t)k * Nn + n];
}
__global__ void k_bias_sum(const float* __restrict__ a, const float* __restrict__ b,
                           float* __restrict__ d, int n) {
  int i = blockIdx.x * blockDim.x + threadIdx.x;
  if (i < n) d[i] = a[i] + b[i];
}
__global__ void k_build_feats(const float* __restrict__ h2, const float* __restrict__ x,
                              h16* __restrict__ feats) {
  long long i = (long long)blockIdx.x * blockDim.x + threadIdx.x;
  if (i >= (long long)NNODE * 66) return;
  int n = (int)(i / 66), j = (int)(i % 66);
  int g = n / NPER, rem = n % NPER, t = rem / KJ, kj = rem % KJ;
  size_t row = (size_t)g * TT + t;
  size_t col = (size_t)kj * 66 + j;
  feats[row * IN1P + col] = (j < HH) ? (h16)h2[(size_t)n * HH + j]
                                     : (h16)x[(size_t)n * 2 + (j - HH)];
}

// ---------------- WMMA GEMM: out[M,N] = A[M,Kp] * Wt[N,Kp]^T + bias ----------------
// Block: 256 threads = 8 waves, tile 32M x 256N (wave = 16M x 64N = 4 D-tiles).
// A 32xKC tile is double-buffered in LDS via async global->LDS copies (ASYNCcnt),
// overlapping the next chunk's copy with this chunk's WMMAs. K % KC == 0 required.
__global__ void k_wmma_gemm(const h16* __restrict__ A, int lda,
                            const h16* __restrict__ Wt, int ldw,
                            const float* __restrict__ bias,
                            float* __restrict__ outF, h16* __restrict__ outH, int ldo,
                            int K, int relu) {
  __shared__ h16 atile[2][32][KC];   // 8 KB double-buffered A stage
  const int tid  = threadIdx.x;
  const int lane = tid & 31;
  const int w    = tid >> 5;
  const int wrow = w >> 2, wcol = w & 3;
  const int mblk = blockIdx.y * 32;
  const int m0   = mblk + wrow * 16;
  const int n0   = blockIdx.x * 256 + wcol * 64;
  const int l15  = lane & 15;
  const int half8 = (lane >> 4) * 8;
  // async stage assignment: each thread copies one 16B chunk of the 32xKC tile
  const int srow   = tid >> 3;        // 0..31
  const int schunk = (tid & 7) * 8;   // half-offset 0..56
  const h16* agrow = A + (size_t)(mblk + srow) * lda + schunk;

  async_copy_b128(agrow, &atile[0][srow][schunk]);   // preload chunk 0
  wait_asynccnt0();
  __syncthreads();

  v8f acc[4];
#pragma unroll
  for (int i = 0; i < 4; ++i) acc[i] = {};
  const int nsteps = K / KC;
  for (int s = 0; s < nsteps; ++s) {
    const int cur = s & 1;
    if (s + 1 < nsteps)
      async_copy_b128(agrow + (size_t)(s + 1) * KC, &atile[cur ^ 1][srow][schunk]);
    const h16* arow = &atile[cur][wrow * 16 + l15][0];
#pragma unroll
    for (int kk = 0; kk < KC; kk += 32) {
      v16h a = make_op(arow + kk + half8, arow + kk + 16 + half8);
#pragma unroll
      for (int nt = 0; nt < 4; ++nt) {
        const h16* brow = Wt + (size_t)(n0 + nt * 16 + l15) * ldw + (size_t)s * KC;
        v16h b = make_op(brow + kk + half8, brow + kk + 16 + half8);
        acc[nt] = wmma32(a, b, acc[nt]);
      }
    }
    wait_asynccnt0();
    __syncthreads();
  }

  const int mhi = (lane >> 4) * 8;
#pragma unroll
  for (int nt = 0; nt < 4; ++nt) {
    int col = n0 + nt * 16 + l15;
    float bv = bias ? bias[col] : 0.f;
#pragma unroll
    for (int r = 0; r < 8; ++r) {
      int row = m0 + r + mhi;
      float v = acc[nt][r] + bv;
      if (relu) v = fmaxf(v, 0.f);
      if (outF) outF[(size_t)row * ldo + col] = v;
      else      outH[(size_t)row * ldo + col] = (h16)v;
    }
  }
}

// ---------------- persistent LSTM recurrence (one kernel per layer,direction) ----------------
// grid = 16 blocks, each owns 16 batch rows; h(f16)/c(f32)/gate-buffer live in LDS.
// Per step: G[16,2048] = h @ Whh^T via v_wmma, then fused gate math.
#define LSTM_SMEM (16*GATE*4 + 16*LHID*4 + 16*LHID*2)
__global__ void k_lstm_dir(const float* __restrict__ Z /*[MTOT][2048], includes x*Wih+biases*/,
                           const h16* __restrict__ Whh /*[2048][512]*/,
                           h16* __restrict__ out /*[MTOT][1024]*/,
                           int outOff, int backward) {
  extern __shared__ unsigned char smem_raw[];
  float* gbuf = (float*)smem_raw;                               // 16 x 2048
  float* cbuf = (float*)(smem_raw + 16 * GATE * 4);             // 16 x 512
  h16*   hbuf = (h16*)(smem_raw + 16 * GATE * 4 + 16 * LHID * 4); // 16 x 512
  const int tid  = threadIdx.x;
  const int lane = tid & 31;
  const int w    = tid >> 5;          // 8 waves, each covers 256 gate columns
  const int l15  = lane & 15;
  const int half8 = (lane >> 4) * 8;
  const int mhi   = (lane >> 4) * 8;
  const int bbase = blockIdx.x * 16;

  for (int i = tid; i < 16 * LHID; i += 256) { cbuf[i] = 0.f; hbuf[i] = (h16)0.f; }
  __syncthreads();

  for (int s = 0; s < TT; ++s) {
    const int t = backward ? (TT - 1 - s) : s;
    const h16* arow = hbuf + l15 * LHID;
#pragma unroll 1
    for (int nt = 0; nt < 16; ++nt) {
      const int n0 = w * 256 + nt * 16;
      const h16* brow = Whh + (size_t)(n0 + l15) * LHID;
      v8f acc = {};
      for (int k0 = 0; k0 < LHID; k0 += 32) {
        v16h a = make_op(arow + k0 + half8, arow + k0 + 16 + half8);
        v16h b = make_op(brow + k0 + half8, brow + k0 + 16 + half8);
        acc = wmma32(a, b, acc);
      }
#pragma unroll
      for (int r = 0; r < 8; ++r)
        gbuf[(r + mhi) * GATE + n0 + l15] = acc[r];
    }
    __syncthreads();
    // fused gates: z = G + Z;  torch order i,f,g,o
    for (int idx = tid; idx < 16 * LHID; idx += 256) {
      const int m = idx >> 9, j = idx & 511;
      const size_t zrow = (size_t)(bbase + m) * TT + t;
      const float* zp = Z + zrow * GATE;
      const float* gp = gbuf + m * GATE;
      float zi = gp[j]            + zp[j];
      float zf = gp[LHID + j]     + zp[LHID + j];
      float zg = gp[2 * LHID + j] + zp[2 * LHID + j];
      float zo = gp[3 * LHID + j] + zp[3 * LHID + j];
      float c = sigf(zf) * cbuf[idx] + sigf(zi) * tanhf(zg);
      float h = sigf(zo) * tanhf(c);
      cbuf[idx] = c;
      hbuf[idx] = (h16)h;
      out[zrow * (2 * LHID) + outOff + j] = (h16)h;
    }
    __syncthreads();
  }
}

// ---------------- small fp32 head2 ----------------
__global__ void k_head2(const h16* __restrict__ r1, const float* __restrict__ w2,
                        const float* __restrict__ b2, float* __restrict__ out) {
  long long i = (long long)blockIdx.x * blockDim.x + threadIdx.x;
  if (i >= (long long)MTOT * OUTD) return;
  int row = (int)(i / OUTD), o = (int)(i % OUTD);
  const h16* rp = r1 + (size_t)row * 256;
  float s = b2[o];
#pragma unroll 8
  for (int k = 0; k < 256; ++k) s += (float)rp[k] * w2[(size_t)k * OUTD + o];
  out[i] = s;
}

// ---------------- host ----------------
static inline size_t alignup(size_t x) { return (x + 255) & ~(size_t)255; }
static inline int cdiv(long long a, int b) { return (int)((a + b - 1) / b); }

extern "C" void kernel_launch(void* const* d_in, const int* in_sizes, int n_in,
                              void* d_out, int out_size, void* d_ws, size_t ws_size,
                              hipStream_t stream) {
  const float* x        = (const float*)d_in[0];
  const int*   ei       = (const int*)d_in[1];
  const float* g1W      = (const float*)d_in[2];
  const float* g1as     = (const float*)d_in[3];
  const float* g1ad     = (const float*)d_in[4];
  const float* bn1g     = (const float*)d_in[6];
  const float* bn1b     = (const float*)d_in[7];
  const float* g2W      = (const float*)d_in[8];
  const float* g2as     = (const float*)d_in[9];
  const float* g2ad     = (const float*)d_in[10];
  const float* bn2g     = (const float*)d_in[12];
  const float* bn2b     = (const float*)d_in[13];
  const float* l1f_wih  = (const float*)d_in[14];
  const float* l1f_whh  = (const float*)d_in[15];
  const float* l1f_bih  = (const float*)d_in[16];
  const float* l1f_bhh  = (const float*)d_in[17];
  const float* l1b_wih  = (const float*)d_in[18];
  const float* l1b_whh  = (const float*)d_in[19];
  const float* l1b_bih  = (const float*)d_in[20];
  const float* l1b_bhh  = (const float*)d_in[21];
  const float* l2f_wih  = (const float*)d_in[22];
  const float* l2f_whh  = (const float*)d_in[23];
  const float* l2f_bih  = (const float*)d_in[24];
  const float* l2f_bhh  = (const float*)d_in[25];
  const float* l2b_wih  = (const float*)d_in[26];
  const float* l2b_whh  = (const float*)d_in[27];
  const float* l2b_bih  = (const float*)d_in[28];
  const float* l2b_bhh  = (const float*)d_in[29];
  const float* h1_w     = (const float*)d_in[30];
  const float* h1_b     = (const float*)d_in[31];
  const float* h2_w     = (const float*)d_in[32];
  const float* h2_b     = (const float*)d_in[33];
  float* out = (float*)d_out;

  const int E = in_sizes[1] / 2;
  const int* esrcIdx = ei;
  const int* edstIdx = ei + E;

  // ---- workspace arena ----
  unsigned char* base = (unsigned char*)d_ws;
  size_t off = 0;
  auto alloc = [&](size_t bytes) -> void* { void* r = base + off; off = alignup(off + bytes); return r; };

  float*    bufA  = (float*)alloc((size_t)NNODE * HH * 4);   // xp (projected features)
  float*    bufB  = (float*)alloc((size_t)NNODE * HH * 4);   // aggregated GAT out (atomic)
  float*    hbn   = (float*)alloc((size_t)NNODE * HH * 4);   // BN+ReLU output
  float*    esrc  = (float*)alloc((size_t)NNODE * 4);
  float*    edst  = (float*)alloc((size_t)NNODE * 4);
  float*    den   = (float*)alloc((size_t)NNODE * 4);
  float*    eself = (float*)alloc((size_t)NNODE * 4);
  unsigned* maxb  = (unsigned*)alloc((size_t)NNODE * 4);
  float*    ebuf  = (float*)alloc((size_t)E * 4);
  float*    stats = (float*)alloc(2 * HH * 4);
  h16*      feats = (h16*)alloc((size_t)MTOT * IN1P * 2);
  h16*      w1f   = (h16*)alloc((size_t)GATE * IN1P * 2);
  h16*      w1b   = (h16*)alloc((size_t)GATE * IN1P * 2);
  h16*      whh1f = (h16*)alloc((size_t)GATE * LHID * 2);
  h16*      whh1b = (h16*)alloc((size_t)GATE * LHID * 2);
  h16*      w2f   = (h16*)alloc((size_t)GATE * 2 * LHID * 2);
  h16*      w2b   = (h16*)alloc((size_t)GATE * 2 * LHID * 2);
  h16*      whh2f = (h16*)alloc((size_t)GATE * LHID * 2);
  h16*      whh2b = (h16*)alloc((size_t)GATE * LHID * 2);
  h16*      h1wT  = (h16*)alloc((size_t)256 * 1024 * 2);
  float*    bs1f  = (float*)alloc(GATE * 4);
  float*    bs1b  = (float*)alloc(GATE * 4);
  float*    bs2f  = (float*)alloc(GATE * 4);
  float*    bs2b  = (float*)alloc(GATE * 4);
  float*    Zf    = (float*)alloc((size_t)MTOT * GATE * 4);
  float*    Zb    = (float*)alloc((size_t)MTOT * GATE * 4);
  h16*      o1    = (h16*)alloc((size_t)MTOT * 2 * LHID * 2);
  h16*      o2    = (h16*)alloc((size_t)MTOT * 2 * LHID * 2);
  h16*      r1    = (h16*)alloc((size_t)MTOT * 256 * 2);

  const int TPB = 256;
  const long long NHtot = (long long)NNODE * HH;

  // ---- weight / bias format conversion (f32 -> f16, [N][K] padded) ----
  {
    long long t1 = (long long)GATE * IN1P;
    k_cvt_pad<<<cdiv(t1, TPB), TPB, 0, stream>>>(l1f_wih, w1f, IN1, IN1P, t1);
    k_cvt_pad<<<cdiv(t1, TPB), TPB, 0, stream>>>(l1b_wih, w1b, IN1, IN1P, t1);
    long long t2 = (long long)GATE * LHID;
    k_cvt_pad<<<cdiv(t2, TPB), TPB, 0, stream>>>(l1f_whh, whh1f, LHID, LHID, t2);
    k_cvt_pad<<<cdiv(t2, TPB), TPB, 0, stream>>>(l1b_whh, whh1b, LHID, LHID, t2);
    long long t3 = (long long)GATE * 2 * LHID;
    k_cvt_pad<<<cdiv(t3, TPB), TPB, 0, stream>>>(l2f_wih, w2f, 2 * LHID, 2 * LHID, t3);
    k_cvt_pad<<<cdiv(t3, TPB), TPB, 0, stream>>>(l2b_wih, w2b, 2 * LHID, 2 * LHID, t3);
    k_cvt_pad<<<cdiv(t2, TPB), TPB, 0, stream>>>(l2f_whh, whh2f, LHID, LHID, t2);
    k_cvt_pad<<<cdiv(t2, TPB), TPB, 0, stream>>>(l2b_whh, whh2b, LHID, LHID, t2);
    k_cvt_T<<<cdiv((long long)256 * 1024, TPB), TPB, 0, stream>>>(h1_w, h1wT, 1024, 256);
    k_bias_sum<<<cdiv(GATE, TPB), TPB, 0, stream>>>(l1f_bih, l1f_bhh, bs1f, GATE);
    k_bias_sum<<<cdiv(GATE, TPB), TPB, 0, stream>>>(l1b_bih, l1b_bhh, bs1b, GATE);
    k_bias_sum<<<cdiv(GATE, TPB), TPB, 0, stream>>>(l2f_bih, l2f_bhh, bs2f, GATE);
    k_bias_sum<<<cdiv(GATE, TPB), TPB, 0, stream>>>(l2b_bih, l2b_bhh, bs2b, GATE);
  }

  // ---- GAT layer (shared launcher): xp in bufA -> bufB -> BN/ReLU into dst ----
  auto gat_layer = [&](const float* asrcW, const float* adstW,
                       const float* bng, const float* bnb, float* dst) {
    (void)hipMemsetAsync(bufB, 0, (size_t)NNODE * HH * 4, stream);
    (void)hipMemsetAsync(maxb, 0, (size_t)NNODE * 4, stream);   // encoded 0 == -inf
    (void)hipMemsetAsync(den, 0, (size_t)NNODE * 4, stream);
    (void)hipMemsetAsync(stats, 0, 2 * HH * 4, stream);
    k_node_dots<<<cdiv(NNODE, TPB), TPB, 0, stream>>>(bufA, asrcW, adstW, esrc, edst);
    k_edge_logit_max<<<cdiv(E, TPB), TPB, 0, stream>>>(esrcIdx, edstIdx, esrc, edst, ebuf, maxb, E);
    k_self_logit_max<<<cdiv(NNODE, TPB), TPB, 0, stream>>>(esrc, edst, eself, maxb);
    k_edge_den<<<cdiv(E, TPB), TPB, 0, stream>>>(edstIdx, ebuf, maxb, den, E);
    k_self_den<<<cdiv(NNODE, TPB), TPB, 0, stream>>>(eself, maxb, den);
    k_edge_scatter<<<cdiv((long long)E * HH, TPB), TPB, 0, stream>>>(
        esrcIdx, edstIdx, ebuf, maxb, den, bufA, bufB, (long long)E * HH);
    k_self_scatter<<<cdiv(NHtot, TPB), TPB, 0, stream>>>(eself, maxb, den, bufA, bufB);
    // GAT bias cancels exactly through BN mean subtraction -> skipped
    k_bn_stats<<<256, dim3(HH, 4), 0, stream>>>(bufB, stats);
    k_bn_apply_relu<<<cdiv(NHtot, TPB), TPB, 0, stream>>>(bufB, stats, bng, bnb, dst);
  };

  // layer 1: x[N,2] -> bufA -> hbn
  k_xp_first<<<cdiv(NHtot, TPB), TPB, 0, stream>>>(x, g1W, bufA);
  gat_layer(g1as, g1ad, bn1g, bn1b, hbn);
  // layer 2: hbn -> bufA(xp) -> hbn (final h2)
  k_xp_dense64<<<cdiv(NHtot, TPB), TPB, 0, stream>>>(hbn, g2W, bufA);
  gat_layer(g2as, g2ad, bn2g, bn2b, hbn);

  // ---- feats: concat(h2, x) -> [MTOT, 1152] f16 (zero padded) ----
  (void)hipMemsetAsync(feats, 0, (size_t)MTOT * IN1P * 2, stream);
  k_build_feats<<<cdiv((long long)NNODE * 66, TPB), TPB, 0, stream>>>(hbn, x, feats);

  // ---- BiLSTM layer 1 ----
  {
    dim3 gz(GATE / 256, MTOT / 32);
    k_wmma_gemm<<<gz, 256, 0, stream>>>(feats, IN1P, w1f, IN1P, bs1f, Zf, nullptr, GATE, IN1P, 0);
    k_wmma_gemm<<<gz, 256, 0, stream>>>(feats, IN1P, w1b, IN1P, bs1b, Zb, nullptr, GATE, IN1P, 0);
    k_lstm_dir<<<16, 256, LSTM_SMEM, stream>>>(Zf, whh1f, o1, 0, 0);
    k_lstm_dir<<<16, 256, LSTM_SMEM, stream>>>(Zb, whh1b, o1, LHID, 1);
  }
  // ---- BiLSTM layer 2 (reuse Zf/Zb) ----
  {
    dim3 gz(GATE / 256, MTOT / 32);
    k_wmma_gemm<<<gz, 256, 0, stream>>>(o1, 2 * LHID, w2f, 2 * LHID, bs2f, Zf, nullptr, GATE, 2 * LHID, 0);
    k_wmma_gemm<<<gz, 256, 0, stream>>>(o1, 2 * LHID, w2b, 2 * LHID, bs2b, Zb, nullptr, GATE, 2 * LHID, 0);
    k_lstm_dir<<<16, 256, LSTM_SMEM, stream>>>(Zf, whh2f, o2, 0, 0);
    k_lstm_dir<<<16, 256, LSTM_SMEM, stream>>>(Zb, whh2b, o2, LHID, 1);
  }
  // ---- head: relu(o2 @ h1_w + h1_b) @ h2_w + h2_b ----
  {
    dim3 gh(1, MTOT / 32);   // N = 256
    k_wmma_gemm<<<gh, 256, 0, stream>>>(o2, 2 * LHID, h1wT, 2 * LHID, h1_b, nullptr, r1, 256, 2 * LHID, 1);
    k_head2<<<cdiv((long long)MTOT * OUTD, TPB), TPB, 0, stream>>>(r1, h2_w, h2_b, out);
  }
  (void)n_in; (void)out_size; (void)ws_size;
}
